// GATv2_66511863546570
// MI455X (gfx1250) — compile-verified, matrix operand order checked by reading, not run on hardware
//
#include <hip/hip_runtime.h>
#include <hip/hip_bf16.h>
#include <math.h>

// ---------------------------------------------------------------------------
// GATv2 2-layer forward for MI455X (gfx1250, wave32).
// GEMMs via v_wmma_f32_16x16x32_bf16 (f32 accum) with pre-swizzled fragment
// layouts (B packed in fragment order in global, A tile staged in fragment
// order in LDS). Edge softmax via ordered-uint atomicMax + f32 atomicAdd
// scatters; node-feature matrices are L2-resident (51MB / 128MB << 192MB L2).
// ---------------------------------------------------------------------------

typedef __attribute__((ext_vector_type(16))) __bf16 v16bf;
typedef __attribute__((ext_vector_type(8)))  float  v8f;

#define F_IN 256
#define D1   128   // [xl1 | xr1]
#define D2   320   // [xl2 | xr2]
#define H1_  8
#define C1_  8
#define H2_  4
#define C2_  40
#define NCLS 40

// monotone float <-> uint mapping so unsigned atomicMax == float max
static __device__ __forceinline__ unsigned f32_ord(float f) {
  unsigned u = __float_as_uint(f);
  return (u & 0x80000000u) ? ~u : (u | 0x80000000u);
}
static __device__ __forceinline__ float ord_f32(unsigned u) {
  unsigned v = (u & 0x80000000u) ? (u & 0x7FFFFFFFu) : ~u;
  return __uint_as_float(v);
}

// ---------------------------------------------------------------------------
__global__ void zero_f32(float* p, long n) {
  long i = (long)blockIdx.x * blockDim.x + threadIdx.x;
  long stride = (long)gridDim.x * blockDim.x;
  for (; i < n; i += stride) p[i] = 0.f;
}

// ---------------------------------------------------------------------------
// Pack [Wl|Wr] -> bf16 weights in WMMA *fragment order*:
//   Bp[((nt*KS + ks)*32 + lane)*16 + e]  holds  W[kk][nt*16 + lane%16]
//   with kk = ks*32 + e + 8*(lane/16) + (e>=8 ? 8 : 0)   (ISA 7.12.2)
// so each lane's 16-element fragment is one contiguous 32B read.
// ---------------------------------------------------------------------------
__global__ void pack_weights(const float* __restrict__ Wl1, const float* __restrict__ Wr1,
                             const float* __restrict__ bl1, const float* __restrict__ br1,
                             const float* __restrict__ Wl2, const float* __restrict__ Wr2,
                             const float* __restrict__ bl2, const float* __restrict__ br2,
                             __bf16* __restrict__ w1p, __bf16* __restrict__ w2p,
                             float* __restrict__ bias1, float* __restrict__ bias2) {
  int i = blockIdx.x * blockDim.x + threadIdx.x;
  const int W1E = F_IN * D1;          // 32768 = 8 ntiles * 8 ksteps * 512
  const int W2E = 64 * D2;            // 20480 = 20 ntiles * 2 ksteps * 512
  if (i < W1E) {
    int b = i >> 9, rem = i & 511;
    int lane = rem >> 4, e = rem & 15;
    int nt = b >> 3, ks = b & 7;                    // KS = 256/32 = 8
    int m = lane & 15, half = lane >> 4;
    int n  = nt * 16 + m;
    int kk = ks * 32 + e + 8 * half + ((e >= 8) ? 8 : 0);
    float v = (n < 64) ? Wl1[kk * 64 + n] : Wr1[kk * 64 + (n - 64)];
    w1p[i] = (__bf16)v;
  } else if (i < W1E + W2E) {
    int t = i - W1E;
    int b = t >> 9, rem = t & 511;
    int lane = rem >> 4, e = rem & 15;
    int nt = b >> 1, ks = b & 1;                    // KS = 64/32 = 2
    int m = lane & 15, half = lane >> 4;
    int n  = nt * 16 + m;
    int kk = ks * 32 + e + 8 * half + ((e >= 8) ? 8 : 0);
    float v = (n < 160) ? Wl2[kk * 160 + n] : Wr2[kk * 160 + (n - 160)];
    w2p[t] = (__bf16)v;
  } else if (i < W1E + W2E + D1) {
    int j = i - (W1E + W2E);
    bias1[j] = (j < 64) ? bl1[j] : br1[j - 64];
  } else if (i < W1E + W2E + D1 + D2) {
    int j = i - (W1E + W2E + D1);
    bias2[j] = (j < 160) ? bl2[j] : br2[j - 160];
  }
}

// ---------------------------------------------------------------------------
// C[nrows x NC] = act(A[nrows x K] (+biasK)) @ W[K x NC] + biasN
// Block: 256 threads = 8 waves; block owns one 16-row tile; each wave walks
// 16-col tiles. A tile staged f32->bf16 into LDS in fragment order; each
// lane's A/B fragments are contiguous 32B (2x ds_load_b128 / global b128).
// ---------------------------------------------------------------------------
template <int K, int NC, int ELU>
__global__ void gemm_wmma_bf16(const float* __restrict__ A, const v16bf* __restrict__ Bp,
                               const float* __restrict__ biasN, const float* __restrict__ biasK,
                               float* __restrict__ Cout, int nrows) {
  constexpr int KS = K / 32;
  __shared__ alignas(32) __bf16 As[16 * K];        // fragment-ordered A tile

  const int rt  = blockIdx.x;
  const int tid = threadIdx.x;

  // stage A tile into LDS in fragment order:
  //   As[(ks*32 + lane)*16 + e] = A[rt*16 + lane%16][kk(ks,lane,e)]
  for (int i = tid; i < 16 * K; i += 256) {
    int ks = i >> 9, rem = i & 511;
    int lane = rem >> 4, e = rem & 15;
    int m = lane & 15, half = lane >> 4;
    int kk  = ks * 32 + e + 8 * half + ((e >= 8) ? 8 : 0);
    int row = rt * 16 + m;
    float v = 0.f;
    if (row < nrows) {
      v = A[(size_t)row * K + kk];
      if (biasK) v += biasK[kk];
      if (ELU) v = (v > 0.f) ? v : (expf(v) - 1.f);
    }
    As[i] = (__bf16)v;
  }
  __syncthreads();

  const int lane = tid & 31;
  const int wid  = tid >> 5;
  const int m    = lane & 15;
  const int half = lane >> 4;
  const bool fullTile = (rt * 16 + 16 <= nrows);

  for (int nt = wid; nt < NC / 16; nt += 8) {
    v8f acc = {0.f, 0.f, 0.f, 0.f, 0.f, 0.f, 0.f, 0.f};
#pragma unroll
    for (int ks = 0; ks < KS; ++ks) {
      v16bf a = *(const v16bf*)(As + (ks * 32 + lane) * 16);
      v16bf b = Bp[(nt * KS + ks) * 32 + lane];
      acc = __builtin_amdgcn_wmma_f32_16x16x32_bf16(
          /*neg_a=*/false, a, /*neg_b=*/false, b,
          /*c_mod=*/(short)0, acc, /*reuse_a=*/false, /*reuse_b=*/false);
    }
    const int col = nt * 16 + m;
    const float bn = biasN[col];
    if (fullTile) {
#pragma unroll
      for (int r = 0; r < 8; ++r)
        Cout[(size_t)(rt * 16 + r + 8 * half) * NC + col] = acc[r] + bn;
    } else {
#pragma unroll
      for (int r = 0; r < 8; ++r) {
        int row = rt * 16 + r + 8 * half;
        if (row < nrows) Cout[(size_t)row * NC + col] = acc[r] + bn;
      }
    }
  }
}

// ---------------------------------------------------------------------------
// Edge pass A: logits + segment max.  thread = (edge, head); vector loads.
// ---------------------------------------------------------------------------
template <int H, int C>
__global__ void edge_logits_t(const float* __restrict__ xlr, int stride, int xroff,
                              const float* __restrict__ att, const int* __restrict__ ei,
                              int E, int Etot,
                              float* __restrict__ elog, unsigned* __restrict__ mbuf) {
  int t = blockIdx.x * blockDim.x + threadIdx.x;
  if (t >= Etot * H) return;
  int e = t / H, h = t - e * H;
  int src, dst;
  if (e < E) { src = ei[e]; dst = ei[E + e]; } else { src = dst = e - E; }
  const float4* xl = (const float4*)(xlr + (size_t)src * stride + h * C);
  const float4* xr = (const float4*)(xlr + (size_t)dst * stride + xroff + h * C);
  const float4* ah = (const float4*)(att + h * C);
  float acc = 0.f;
#pragma unroll
  for (int c = 0; c < C / 4; ++c) {
    float4 a = xl[c], b = xr[c], w = ah[c];
    float v;
    v = a.x + b.x; v = (v > 0.f) ? v : 0.2f * v; acc += v * w.x;
    v = a.y + b.y; v = (v > 0.f) ? v : 0.2f * v; acc += v * w.y;
    v = a.z + b.z; v = (v > 0.f) ? v : 0.2f * v; acc += v * w.z;
    v = a.w + b.w; v = (v > 0.f) ? v : 0.2f * v; acc += v * w.w;
  }
  elog[t] = acc;
  atomicMax(&mbuf[(size_t)dst * H + h], f32_ord(acc));
}

// Edge pass B: exp + segment sum
template <int H>
__global__ void edge_expsum_t(const float* __restrict__ elog, const unsigned* __restrict__ mbuf,
                              const int* __restrict__ ei, int E, int Etot,
                              float* __restrict__ sbuf) {
  int t = blockIdx.x * blockDim.x + threadIdx.x;
  if (t >= Etot * H) return;
  int e = t / H, h = t - e * H;
  int dst = (e < E) ? ei[E + e] : (e - E);
  float mx = ord_f32(mbuf[(size_t)dst * H + h]);
  atomicAdd(&sbuf[(size_t)dst * H + h], expf(elog[t] - mx));
}

// Edge pass C: alpha-weighted scatter of source messages
template <int H, int C, int CONCAT>
__global__ void edge_scatter_t(const float* __restrict__ elog, const unsigned* __restrict__ mbuf,
                               const float* __restrict__ sbuf,
                               const float* __restrict__ xlr, int stride,
                               const int* __restrict__ ei, int E, int Etot,
                               float* __restrict__ outf) {
  int t = blockIdx.x * blockDim.x + threadIdx.x;
  if (t >= Etot * H) return;
  int e = t / H, h = t - e * H;
  int src, dst;
  if (e < E) { src = ei[e]; dst = ei[E + e]; } else { src = dst = e - E; }
  float mx = ord_f32(mbuf[(size_t)dst * H + h]);
  float s  = sbuf[(size_t)dst * H + h];
  float alpha = expf(elog[t] - mx) / (s + 1e-16f);
  const float4* xl = (const float4*)(xlr + (size_t)src * stride + h * C);
  float* o = CONCAT ? (outf + ((size_t)dst * H + h) * C) : (outf + (size_t)dst * C);
#pragma unroll
  for (int c = 0; c < C / 4; ++c) {
    float4 a = xl[c];
    atomicAdd(&o[4 * c + 0], alpha * a.x);
    atomicAdd(&o[4 * c + 1], alpha * a.y);
    atomicAdd(&o[4 * c + 2], alpha * a.z);
    atomicAdd(&o[4 * c + 3], alpha * a.w);
  }
}

// ---------------------------------------------------------------------------
// head-mean + bias + log_softmax, in place on d_out. One wave32 per node.
// ---------------------------------------------------------------------------
__global__ void final_logsoftmax(float* __restrict__ out, const float* __restrict__ b2, int n) {
  int wid  = threadIdx.x >> 5;
  int lane = threadIdx.x & 31;
  int node = blockIdx.x * (blockDim.x >> 5) + wid;
  if (node >= n) return;
  float* row = out + (size_t)node * NCLS;
  float v0 = -INFINITY, v1 = -INFINITY;
  if (lane < NCLS)      v0 = row[lane]      * 0.25f + b2[lane];        // mean over 4 heads
  if (lane < NCLS - 32) v1 = row[lane + 32] * 0.25f + b2[lane + 32];
  float mx = fmaxf(v0, v1);
  for (int o = 16; o > 0; o >>= 1) mx = fmaxf(mx, __shfl_xor(mx, o, 32));
  float s = 0.f;
  if (lane < NCLS)      s += expf(v0 - mx);
  if (lane < NCLS - 32) s += expf(v1 - mx);
  for (int o = 16; o > 0; o >>= 1) s += __shfl_xor(s, o, 32);
  float lg = logf(s);
  if (lane < NCLS)      row[lane]      = v0 - mx - lg;
  if (lane < NCLS - 32) row[lane + 32] = v1 - mx - lg;
}

// ---------------------------------------------------------------------------
extern "C" void kernel_launch(void* const* d_in, const int* in_sizes, int n_in,
                              void* d_out, int out_size, void* d_ws, size_t ws_size,
                              hipStream_t stream) {
  const float* x    = (const float*)d_in[0];
  const int*   ei   = (const int*)  d_in[1];
  const float* Wl1  = (const float*)d_in[2];
  const float* bl1  = (const float*)d_in[3];
  const float* Wr1  = (const float*)d_in[4];
  const float* br1  = (const float*)d_in[5];
  const float* att1 = (const float*)d_in[6];
  const float* b1   = (const float*)d_in[7];
  const float* Wl2  = (const float*)d_in[8];
  const float* bl2  = (const float*)d_in[9];
  const float* Wr2  = (const float*)d_in[10];
  const float* br2  = (const float*)d_in[11];
  const float* att2 = (const float*)d_in[12];
  const float* b2   = (const float*)d_in[13];
  float* out = (float*)d_out;

  const int N = in_sizes[0] / F_IN;
  const int E = in_sizes[1] / 2;
  const int Etot = E + N;                         // + self loops

  // ---- workspace carve-up ----
  char* ws = (char*)d_ws;
  size_t off = 0;
  auto carve = [&](size_t bytes) -> char* {
    char* p = ws + off;
    off += (bytes + 255) & ~(size_t)255;
    return p;
  };
  __bf16* w1p   = (__bf16*)carve((size_t)F_IN * D1 * sizeof(__bf16));
  __bf16* w2p   = (__bf16*)carve((size_t)64 * D2 * sizeof(__bf16));
  float*  bias1 = (float*) carve(D1 * sizeof(float));
  float*  bias2 = (float*) carve(D2 * sizeof(float));
  float*  xlr1  = (float*) carve((size_t)N * D1 * sizeof(float));
  float*  elog  = (float*) carve((size_t)Etot * H1_ * sizeof(float));   // reused for layer 2
  float*  msh   = (float*) carve((size_t)N * (H1_ + H1_ + 64) * sizeof(float)); // m1|s1|h1
  unsigned* m1  = (unsigned*)msh;
  float*  s1    = msh + (size_t)N * H1_;
  float*  h1    = msh + (size_t)N * H1_ * 2;
  float*  xlr2  = (float*) carve((size_t)N * D2 * sizeof(float));
  float*  ms2   = (float*) carve((size_t)N * (H2_ + H2_) * sizeof(float));      // m2|s2
  unsigned* m2  = (unsigned*)ms2;
  float*  s2    = ms2 + (size_t)N * H2_;

  const int TB = 256;

  // 0) pack weights/biases to bf16 fragment order
  {
    int tot = F_IN * D1 + 64 * D2 + D1 + D2;
    pack_weights<<<(tot + TB - 1) / TB, TB, 0, stream>>>(Wl1, Wr1, bl1, br1,
                                                         Wl2, Wr2, bl2, br2,
                                                         w1p, w2p, bias1, bias2);
  }
  // 1) zero m1|s1|h1, m2|s2, and d_out (accumulator for layer-2 scatter)
  zero_f32<<<2048, TB, 0, stream>>>(msh, (long)N * (H1_ + H1_ + 64));
  zero_f32<<<512,  TB, 0, stream>>>(ms2, (long)N * (H2_ + H2_));
  zero_f32<<<1024, TB, 0, stream>>>(out, (long)N * NCLS);

  // 2) layer-1 GEMM: xlr1 = x @ [Wl1|Wr1] + [bl1|br1]
  gemm_wmma_bf16<F_IN, D1, 0><<<(N + 15) / 16, TB, 0, stream>>>(
      x, (const v16bf*)w1p, bias1, nullptr, xlr1, N);

  // 3) layer-1 attention
  {
    int tot = Etot * H1_, nb = (tot + TB - 1) / TB;
    edge_logits_t<H1_, C1_><<<nb, TB, 0, stream>>>(xlr1, D1, 64, att1, ei, E, Etot, elog, m1);
    edge_expsum_t<H1_>    <<<nb, TB, 0, stream>>>(elog, m1, ei, E, Etot, s1);
    edge_scatter_t<H1_, C1_, 1><<<nb, TB, 0, stream>>>(elog, m1, s1, xlr1, D1, ei, E, Etot, h1);
  }

  // 4) layer-2 GEMM with fused (+b1, ELU) on the A operand:
  //    xlr2 = elu(h1 + b1) @ [Wl2|Wr2] + [bl2|br2]
  gemm_wmma_bf16<64, D2, 1><<<(N + 15) / 16, TB, 0, stream>>>(
      h1, (const v16bf*)w2p, bias2, b1, xlr2, N);

  // 5) layer-2 attention; heads accumulated straight into d_out [N,40]
  {
    int tot = Etot * H2_, nb = (tot + TB - 1) / TB;
    edge_logits_t<H2_, C2_><<<nb, TB, 0, stream>>>(xlr2, D2, 160, att2, ei, E, Etot, elog, m2);
    edge_expsum_t<H2_>    <<<nb, TB, 0, stream>>>(elog, m2, ei, E, Etot, s2);
    edge_scatter_t<H2_, C2_, 0><<<nb, TB, 0, stream>>>(elog, m2, s2, xlr2, D2, ei, E, Etot, out);
  }

  // 6) head mean + bias + log_softmax (in place, wave per node)
  final_logsoftmax<<<(N + 7) / 8, TB, 0, stream>>>(out, b2, N);
}
